// MultiHeadedAttention_6820408066611
// MI455X (gfx1250) — matrix operand order repaired and checked by reading, not compiled, not run on hardware
//
#include <hip/hip_runtime.h>
#include <hip/hip_bf16.h>

// ---------------------------------------------------------------------------
// MHA for MI455X (gfx1250), wave32 + WMMA f32_16x16x32_f16 pipeline,
// async-to-LDS staging (ASYNCcnt) with double buffering.
// Softmax row-sums computed on the matrix pipe (P @ ones) instead of
// lane-butterflies. B=4, S=2048, D=1024, H=16, DK=64.
// ---------------------------------------------------------------------------

#define BB  4
#define SS  2048
#define DD  1024
#define HH  16
#define DKK 64

typedef __attribute__((ext_vector_type(16))) _Float16 v16h;
typedef __attribute__((ext_vector_type(8)))  _Float16 v8h;
typedef __attribute__((ext_vector_type(8)))  float    v8f;

#define WAIT_ASYNC(n) asm volatile("s_wait_asynccnt " #n ::: "memory")

__device__ __forceinline__ v8f vzero8f() {
  v8f z;
#pragma unroll
  for (int i = 0; i < 8; ++i) z[i] = 0.0f;
  return z;
}

__device__ __forceinline__ v16h combine16(v8h lo, v8h hi) {
  v16h r;
#pragma unroll
  for (int i = 0; i < 8; ++i) { r[i] = lo[i]; r[i + 8] = hi[i]; }
  return r;
}

// A-matrix fragment (16x32 f16), CDNA5 layout:
//  lanes 0-15 : M=lane,    K = 0..7 (v0-3) and 16..23 (v4-7)
//  lanes 16-31: M=lane-16, K = 8..15       and 24..31
__device__ __forceinline__ v16h load_a_frag(const _Float16* base, int stride) {
  const int lane = threadIdx.x & 31;
  const _Float16* p = base + (lane & 15) * stride + ((lane < 16) ? 0 : 8);
  return combine16(*(const v8h*)p, *(const v8h*)(p + 16));
}

// B-matrix fragment (32x16 f16): lane holds column N=lane%16, K as A-frag:
//  lanes 0-15 : K=0..7,16..23 ; lanes 16-31 : K=8..15,24..31
// base points at K-contiguous (transposed) storage: tileT[N][K].
__device__ __forceinline__ v16h load_b_frag(const _Float16* base, int stride) {
  const int lane = threadIdx.x & 31;
  const _Float16* p = base + (lane & 15) * stride + ((lane < 16) ? 0 : 8);
  return combine16(*(const v8h*)p, *(const v8h*)(p + 16));
}

__device__ __forceinline__ v8f wmma_f16(v16h a, v16h b, v8f c) {
  return __builtin_amdgcn_wmma_f32_16x16x32_f16(
      /*neg_a=*/false, a, /*neg_b=*/false, b,
      /*c_mod=*/(short)0, c, /*reuse_a=*/false, /*reuse_b=*/false);
}

// Async global->LDS 16B copy (CDNA5 ASYNCcnt path). The LDS destination
// address is the wave-relative low 32 bits of the flat shared pointer
// (the VGPR offset is added to the LDS base, per ISA 08_async_tensor).
__device__ __forceinline__ void async_copy16(_Float16* lds_dst,
                                             const _Float16* gsrc) {
  const unsigned loff = (unsigned)(uintptr_t)lds_dst;
  asm volatile("global_load_async_to_lds_b128 %0, %1, off"
               :
               : "v"(loff), "v"((unsigned long long)(uintptr_t)gsrc)
               : "memory");
}

// ---------------------------------------------------------------------------
// Elementwise f32 -> f16 cast
// ---------------------------------------------------------------------------
__global__ void cast_f32_f16(const float* __restrict__ in,
                             _Float16* __restrict__ out, int n) {
  for (int i = blockIdx.x * blockDim.x + threadIdx.x; i < n;
       i += gridDim.x * blockDim.x)
    out[i] = (_Float16)in[i];
}

// ---------------------------------------------------------------------------
// Weight transpose + cast: Wt[n*Dd + k] = (f16) W[k*Dd + n]
// ---------------------------------------------------------------------------
__global__ void transpose_cast(const float* __restrict__ W,
                               _Float16* __restrict__ Wt, int Dd) {
  __shared__ float tile[32][33];
  const int bn = blockIdx.x * 32;  // n dimension
  const int bk = blockIdx.y * 32;  // k dimension
#pragma unroll
  for (int i = threadIdx.y; i < 32; i += 8)
    tile[i][threadIdx.x] = W[(size_t)(bk + i) * Dd + bn + threadIdx.x];
  __syncthreads();
#pragma unroll
  for (int i = threadIdx.y; i < 32; i += 8)
    Wt[(size_t)(bn + i) * Dd + bk + threadIdx.x] =
        (_Float16)tile[threadIdx.x][i];
}

// ---------------------------------------------------------------------------
// GEMM: out = A[MxK](f16) * Wt[NxK](f16)^T + bias
// Block tile 256x128, 8 waves in 4(M)x2(N), wave tile 64x64 (4x4 WMMA).
// Double-buffered LDS (60 KB), async-to-LDS staging (6 copies/thread/tile).
// MODE 0: f16 row-major [M,N]; MODE 1: f16 head-split [B,H,S,DK] (*scale);
// MODE 2: f32 row-major [M,N].
// ---------------------------------------------------------------------------
#define GEMM_LDSS 40  // LDS row stride in halves (80B = 5x16B, aligned rows)

template <int MODE>
__global__ __launch_bounds__(256) void gemm_f16_wmma(
    const _Float16* __restrict__ A, const _Float16* __restrict__ Bt,
    const float* __restrict__ bias, void* __restrict__ out,
    int M, int N, int K, float scale) {
  __shared__ __align__(16) _Float16 Als[2][256 * GEMM_LDSS];  // 2 x 20 KB
  __shared__ __align__(16) _Float16 Bls[2][128 * GEMM_LDSS];  // 2 x 10 KB

  const int tid = threadIdx.x;
  const int lane = tid & 31, wave = tid >> 5;
  const int wm = wave & 3, wn = wave >> 2;  // 4 x 2 wave grid
  const int bm = blockIdx.y * 256, bn = blockIdx.x * 128;

  v8f acc[4][4];
#pragma unroll
  for (int mt = 0; mt < 4; ++mt)
#pragma unroll
    for (int nt = 0; nt < 4; ++nt) acc[mt][nt] = vzero8f();

  // A tile 256x32: 1024 16B-chunks -> 4/thread; B tile 128x32: 512 -> 2/thread
  auto stage = [&](int buf, int k0) {
#pragma unroll
    for (int i = 0; i < 4; ++i) {
      const int c = tid + i * 256;
      const int row = c >> 2, kc = (c & 3) << 3;
      async_copy16(&Als[buf][row * GEMM_LDSS + kc],
                   &A[(size_t)(bm + row) * K + k0 + kc]);
    }
#pragma unroll
    for (int i = 0; i < 2; ++i) {
      const int c = tid + i * 256;
      const int row = c >> 2, kc = (c & 3) << 3;
      async_copy16(&Bls[buf][row * GEMM_LDSS + kc],
                   &Bt[(size_t)(bn + row) * K + k0 + kc]);
    }
  };

  stage(0, 0);  // prologue: tile 0 in flight

  int buf = 0;
  for (int k0 = 0; k0 < K; k0 += 32, buf ^= 1) {
    if (k0 + 32 < K) {
      stage(buf ^ 1, k0 + 32);  // keep next tile in flight
      WAIT_ASYNC(6);            // 6 newest outstanding = current tile resident
    } else {
      WAIT_ASYNC(0);
    }
    __syncthreads();

    v16h af[4], bf[4];
#pragma unroll
    for (int mt = 0; mt < 4; ++mt)
      af[mt] =
          load_a_frag(&Als[buf][(wm * 64 + mt * 16) * GEMM_LDSS], GEMM_LDSS);
#pragma unroll
    for (int nt = 0; nt < 4; ++nt)
      bf[nt] =
          load_b_frag(&Bls[buf][(wn * 64 + nt * 16) * GEMM_LDSS], GEMM_LDSS);
#pragma unroll
    for (int mt = 0; mt < 4; ++mt)
#pragma unroll
      for (int nt = 0; nt < 4; ++nt)
        acc[mt][nt] = wmma_f16(af[mt], bf[nt], acc[mt][nt]);
    __syncthreads();  // protect buf^1 before next stage() overwrites it
  }

  // Epilogue: C/D layout -> lane holds col n0, rows r (lanes 0-15) / 8+r.
#pragma unroll
  for (int mt = 0; mt < 4; ++mt) {
#pragma unroll
    for (int nt = 0; nt < 4; ++nt) {
      const int n0 = bn + wn * 64 + nt * 16 + (lane & 15);
      const float bias_v = bias[n0];
#pragma unroll
      for (int r = 0; r < 8; ++r) {
        const int m = bm + wm * 64 + mt * 16 + ((lane < 16) ? r : 8 + r);
        const float v = acc[mt][nt][r] + bias_v;
        if (MODE == 1) {
          const int h = n0 >> 6, dk = n0 & 63;  // DK = 64
          const int b = m >> 11, s = m & 2047;  // S  = 2048
          ((_Float16*)out)[(((size_t)(b * HH + h) * SS + s) << 6) + dk] =
              (_Float16)(v * scale);
        } else if (MODE == 0) {
          ((_Float16*)out)[(size_t)m * N + n0] = (_Float16)v;
        } else {
          ((float*)out)[(size_t)m * N + n0] = v;
        }
      }
    }
  }
}

// ---------------------------------------------------------------------------
// Flash attention: one block = one (b,h) head, 128 query rows (8 waves x 16).
// 64-key tiles: K tile (64x64) async-staged row-major; V tile staged
// transposed (Vt[dk][key], stride 72). P re-laid out via per-wave LDS.
// 18 WMMA per iteration (8 scores + 2 row-sum + 8 PV); only the row-max
// needs a lane butterfly. Q pre-scaled by 1/sqrt(DK) at projection time.
// ---------------------------------------------------------------------------
#define VT_STRIDE 72  // halves, 144B = 9x16B rows (16B-aligned)

__global__ __launch_bounds__(256) void attn_wmma(
    const _Float16* __restrict__ Qh, const _Float16* __restrict__ Kh,
    const _Float16* __restrict__ Vh, _Float16* __restrict__ Ctx) {
  __shared__ __align__(16) _Float16 Kls[64 * 64];          // [key][dk]  8 KB
  __shared__ __align__(16) _Float16 Vtls[64 * VT_STRIDE];  // [dk][key]  9 KB
  __shared__ __align__(16) _Float16 Pls[8][16 * 64];       // per-wave  16 KB

  const int tid = threadIdx.x, lane = tid & 31, wave = tid >> 5;
  const int bh = blockIdx.y;                    // b*H + h
  const int q0 = blockIdx.x * 128 + wave * 16;  // first query row of wave
  const size_t headBase = (size_t)bh * SS * DKK;
  const _Float16* qp = Qh + headBase + (size_t)q0 * DKK;

  // Q fragments live in registers for the whole pass (dk 0..31 and 32..63).
  const v16h aq0 = load_a_frag(qp, DKK);
  const v16h aq1 = load_a_frag(qp + 32, DKK);

  // Constant all-ones B fragment: rowsum(P) = P @ ones on the matrix pipe.
  v16h vones;
#pragma unroll
  for (int i = 0; i < 16; ++i) vones[i] = (_Float16)1.0f;

  v8f acc[4];
#pragma unroll
  for (int ng = 0; ng < 4; ++ng) acc[ng] = vzero8f();
  float rm[8], rl[8];
#pragma unroll
  for (int r = 0; r < 8; ++r) { rm[r] = -1e30f; rl[r] = 0.0f; }

  const int vkey = tid & 63, vdg = tid >> 6;  // V-stage: key, dk-group (0..3)

  for (int jt = 0; jt < SS / 64; ++jt) {
    const int jb = jt * 64;
    // stage K tile (async): 64 keys x 64 dk = 512 chunks -> 2 per thread
#pragma unroll
    for (int i = 0; i < 2; ++i) {
      const int c = tid + i * 256;
      const int key = c >> 3, kc = (c & 7) << 3;
      async_copy16(&Kls[key * 64 + kc],
                   (const _Float16*)(Kh + headBase +
                                     (size_t)(jb + key) * DKK + kc));
    }
    // stage V tile transposed: Vt[dk][key]; 16 halves per thread
#pragma unroll
    for (int part = 0; part < 2; ++part) {
      const int dk0 = vdg * 16 + part * 8;
      const v8h v =
          *(const v8h*)(Vh + headBase + (size_t)(jb + vkey) * DKK + dk0);
#pragma unroll
      for (int e = 0; e < 8; ++e) Vtls[(dk0 + e) * VT_STRIDE + vkey] = v[e];
    }
    WAIT_ASYNC(0);
    __syncthreads();

    // scores: four 16x16 tiles (keys jb+16t .. jb+16t+15)
    v8f s[4];
#pragma unroll
    for (int t = 0; t < 4; ++t) {
      v8f st = vzero8f();
      st = wmma_f16(aq0, load_b_frag(Kls + t * 16 * 64, 64), st);
      st = wmma_f16(aq1, load_b_frag(Kls + t * 16 * 64 + 32, 64), st);
      s[t] = st;
    }

    // online softmax; row r lives across 16 lanes of each half-wave
    float al[8];
#pragma unroll
    for (int r = 0; r < 8; ++r) {
      float mx = fmaxf(fmaxf(s[0][r], s[1][r]), fmaxf(s[2][r], s[3][r]));
      mx = fmaxf(mx, __shfl_xor(mx, 1, 16));
      mx = fmaxf(mx, __shfl_xor(mx, 2, 16));
      mx = fmaxf(mx, __shfl_xor(mx, 4, 16));
      mx = fmaxf(mx, __shfl_xor(mx, 8, 16));
      const float mn = fmaxf(rm[r], mx);
      al[r] = __expf(rm[r] - mn);
      rm[r] = mn;
      const int Mrow = (lane < 16) ? r : 8 + r;
#pragma unroll
      for (int t = 0; t < 4; ++t)
        Pls[wave][Mrow * 64 + t * 16 + (lane & 15)] =
            (_Float16)__expf(s[t][r] - mn);
#pragma unroll
      for (int ng = 0; ng < 4; ++ng) acc[ng][r] *= al[r];
    }

    // intra-wave LDS store->load ordering (CDNA5 split DS counter)
    asm volatile("s_wait_dscnt 0x0" ::: "memory");
    const v16h pf0 = load_a_frag(Pls[wave], 64);       // keys jb..jb+31
    const v16h pf1 = load_a_frag(Pls[wave] + 32, 64);  // keys jb+32..jb+63

    // row-sum on the matrix pipe: psum[r] = rowsum(P) replicated per lane
    v8f psum = vzero8f();
    psum = wmma_f16(pf0, vones, psum);
    psum = wmma_f16(pf1, vones, psum);
#pragma unroll
    for (int r = 0; r < 8; ++r) rl[r] = rl[r] * al[r] + psum[r];

#pragma unroll
    for (int ng = 0; ng < 4; ++ng) {
      const _Float16* vb = &Vtls[ng * 16 * VT_STRIDE];
      acc[ng] = wmma_f16(pf0, load_b_frag(vb, VT_STRIDE), acc[ng]);
      acc[ng] = wmma_f16(pf1, load_b_frag(vb + 32, VT_STRIDE), acc[ng]);
    }
    __syncthreads();
  }

  // normalize (reciprocal once per row) and write context [B,S,D] f16
  float inv[8];
#pragma unroll
  for (int r = 0; r < 8; ++r) inv[r] = __builtin_amdgcn_rcpf(rl[r]);
  const int b = bh >> 4, h = bh & 15;
#pragma unroll
  for (int ng = 0; ng < 4; ++ng) {
    const int col = h * 64 + ng * 16 + (lane & 15);
#pragma unroll
    for (int r = 0; r < 8; ++r) {
      const int s = q0 + ((lane < 16) ? r : 8 + r);
      Ctx[(size_t)(b * SS + s) * DD + col] = (_Float16)(acc[ng][r] * inv[r]);
    }
  }
}

// ---------------------------------------------------------------------------
// Host-side orchestration
// ---------------------------------------------------------------------------
extern "C" void kernel_launch(void* const* d_in, const int* in_sizes, int n_in,
                              void* d_out, int out_size, void* d_ws,
                              size_t ws_size, hipStream_t stream) {
  (void)in_sizes; (void)n_in; (void)out_size; (void)ws_size;
  const float* q  = (const float*)d_in[0];
  const float* k  = (const float*)d_in[1];
  const float* v  = (const float*)d_in[2];
  const float* Wq = (const float*)d_in[3];
  const float* bq = (const float*)d_in[4];
  const float* Wk = (const float*)d_in[5];
  const float* bk = (const float*)d_in[6];
  const float* Wv = (const float*)d_in[7];
  const float* bv = (const float*)d_in[8];
  const float* Wo = (const float*)d_in[9];
  const float* bo = (const float*)d_in[10];

  const size_t NX = (size_t)BB * SS * DD;  // 8,388,608 activations
  const size_t NW = (size_t)DD * DD;       // 1,048,576 weights

  char* ws = (char*)d_ws;
  size_t off = 0;
  auto alloc = [&](size_t bytes) -> char* {
    char* p = ws + off;
    off += (bytes + 255) & ~(size_t)255;
    return p;
  };

  _Float16* Xq  = (_Float16*)alloc(NX * 2);
  _Float16* Xk  = (_Float16*)alloc(NX * 2);
  _Float16* Xv  = (_Float16*)alloc(NX * 2);
  _Float16* Wqt = (_Float16*)alloc(NW * 2);
  _Float16* Wkt = (_Float16*)alloc(NW * 2);
  _Float16* Wvt = (_Float16*)alloc(NW * 2);
  _Float16* Wot = (_Float16*)alloc(NW * 2);
  _Float16* Qh  = (_Float16*)alloc(NX * 2);
  _Float16* Kh  = (_Float16*)alloc(NX * 2);
  _Float16* Vh  = (_Float16*)alloc(NX * 2);
  _Float16* Ctx = (_Float16*)alloc(NX * 2);

  // 1) cast activations to f16
  cast_f32_f16<<<2048, 256, 0, stream>>>(q, Xq, (int)NX);
  cast_f32_f16<<<2048, 256, 0, stream>>>(k, Xk, (int)NX);
  cast_f32_f16<<<2048, 256, 0, stream>>>(v, Xv, (int)NX);

  // 2) transpose+cast weights to [N][K] f16
  dim3 tblk(32, 8), tgrd(DD / 32, DD / 32);
  transpose_cast<<<tgrd, tblk, 0, stream>>>(Wq, Wqt, DD);
  transpose_cast<<<tgrd, tblk, 0, stream>>>(Wk, Wkt, DD);
  transpose_cast<<<tgrd, tblk, 0, stream>>>(Wv, Wvt, DD);
  transpose_cast<<<tgrd, tblk, 0, stream>>>(Wo, Wot, DD);

  // 3) QKV projections -> head-split [B,H,S,DK]; Q scaled by 1/sqrt(DK)
  dim3 ggrd(DD / 128, (BB * SS) / 256);  // (8, 32)
  gemm_f16_wmma<1><<<ggrd, 256, 0, stream>>>(Xq, Wqt, bq, Qh, BB * SS, DD, DD,
                                             0.125f);
  gemm_f16_wmma<1><<<ggrd, 256, 0, stream>>>(Xk, Wkt, bk, Kh, BB * SS, DD, DD,
                                             1.0f);
  gemm_f16_wmma<1><<<ggrd, 256, 0, stream>>>(Xv, Wvt, bv, Vh, BB * SS, DD, DD,
                                             1.0f);

  // 4) flash attention -> context [B,S,D] f16
  dim3 agrd(SS / 128, BB * HH);  // (16, 64)
  attn_wmma<<<agrd, 256, 0, stream>>>(Qh, Kh, Vh, Ctx);

  // 5) output projection -> f32 d_out
  gemm_f16_wmma<2><<<ggrd, 256, 0, stream>>>(Ctx, Wot, bo, (float*)d_out,
                                             BB * SS, DD, DD, 1.0f);
}
